// GroupedQueryAttention_83614423318587
// MI455X (gfx1250) — compile-verified, matrix operand order checked by reading, not run on hardware
//
#include <hip/hip_runtime.h>
#include <stdint.h>

// Problem constants (match reference)
#define B_  2
#define S_  2048
#define D_  2048
#define H_  32
#define G_  8
#define HD_ 64

typedef __attribute__((ext_vector_type(16))) __bf16        v16bf;
typedef __attribute__((ext_vector_type(8)))  float         v8f;
typedef __attribute__((ext_vector_type(4)))  unsigned int  u32x4;
typedef __attribute__((ext_vector_type(4)))  int           v4i;

// CDNA5 async global->LDS copy (ASYNCcnt-tracked, bypasses VGPRs).
#if defined(__has_builtin)
#  if __has_builtin(__builtin_amdgcn_global_load_async_to_lds_b128) && \
      __has_builtin(__builtin_amdgcn_s_wait_asynccnt)
#    define HAVE_ASYNC_LDS 1
#  endif
#endif
#ifndef HAVE_ASYNC_LDS
#  define HAVE_ASYNC_LDS 0
#endif

// Builtin signature (from compiler diagnostic): arg0 = v4i in AS(1) (global
// src), arg1 = v4i in AS(3) (LDS dst), then imm offset, imm cpol.
typedef __attribute__((address_space(1))) v4i glb_v4i;
typedef __attribute__((address_space(3))) v4i lds_v4i;

// Generic->AS casts go through integers: on amdgcn the low 32 bits of a
// generic LDS address are the LDS offset (aperture in the high bits), so the
// truncating cast is exact; global generic addresses are the 64-bit VA.
__device__ __forceinline__ glb_v4i* as_glb_v4(const unsigned short* p) {
  return (glb_v4i*)(size_t)p;
}
__device__ __forceinline__ lds_v4i* as_lds_v4(unsigned short* p) {
  return (lds_v4i*)(unsigned int)(size_t)p;
}

union FragU { v16bf bf; u32x4 q[2]; };

__device__ __forceinline__ unsigned short f2bf(float f) {
  unsigned int u = __float_as_uint(f);
  u += 0x7FFFu + ((u >> 16) & 1u);   // round-to-nearest-even
  return (unsigned short)(u >> 16);
}

// Build a 16x32(bf16) A/B fragment: 16 bf16 per lane = two 16B chunks.
// CDNA5 16-bit operand layout: lane half selects K-offset 0/8; elements
// 0..7 hold K = kh..kh+7, elements 8..15 hold K = 16+kh..16+kh+7.
__device__ __forceinline__ v16bf load_frag16(const unsigned short* p0,
                                             const unsigned short* p1) {
  FragU f;
  f.q[0] = *(const u32x4*)p0;
  f.q[1] = *(const u32x4*)p1;
  return f.bf;
}

// ---------------------------------------------------------------------------
// Elementwise fp32 -> bf16 cast
// ---------------------------------------------------------------------------
__global__ void cast_f32_bf16_kernel(const float* __restrict__ in,
                                     unsigned short* __restrict__ out,
                                     size_t n) {
  size_t i = (size_t)blockIdx.x * blockDim.x + threadIdx.x;
  if (i < n) out[i] = f2bf(in[i]);
}

// ---------------------------------------------------------------------------
// Tiled transpose + cast: in fp32 (R x C) -> out bf16 (C x R)
// block (32,8); 32x32 tile through padded LDS.
// ---------------------------------------------------------------------------
__global__ __launch_bounds__(256) void transpose_cast_kernel(
    const float* __restrict__ in, unsigned short* __restrict__ out,
    int R, int C) {
  __shared__ float tile[32][33];
  const int tx = threadIdx.x, ty = threadIdx.y;
  const int r0 = blockIdx.y * 32, c0 = blockIdx.x * 32;
#pragma unroll
  for (int j = 0; j < 4; ++j)
    tile[ty + j * 8][tx] = in[(size_t)(r0 + ty + j * 8) * C + c0 + tx];
  __syncthreads();
#pragma unroll
  for (int j = 0; j < 4; ++j)
    out[(size_t)(c0 + ty + j * 8) * R + r0 + tx] = f2bf(tile[tx][ty + j * 8]);
}

// ---------------------------------------------------------------------------
// bf16 GEMM via v_wmma_f32_16x16x32_bf16, double-buffered LDS with CDNA5
// async global->LDS copies (ASYNCcnt) overlapping tile t+1 fetch with tile t
// WMMAs.
//   A  : M x K (row major, bf16)        -- per-lane row fragments
//   Bt : N x K (row major, bf16) = B^T  -- per-lane column fragments
//   C  : M x N (fp32)
// Block = 256 threads = 8 waves; block tile 128x128; wave tile 32x64; BK=32.
// ---------------------------------------------------------------------------
template <bool NT_STORE>
__global__ __launch_bounds__(256) void gemm_bf16_kernel(
    const unsigned short* __restrict__ A,
    const unsigned short* __restrict__ Bt,
    float* __restrict__ C,
    int M, int N, int K) {
  __shared__ unsigned short sA[2][128 * 32];
  __shared__ unsigned short sB[2][128 * 32];

  const int tid  = threadIdx.x;
  const int lane = tid & 31;
  const int wid  = tid >> 5;        // 0..7
  const int wm   = wid >> 1;        // 0..3 : 32-row slab
  const int wn   = wid & 1;         // 0..1 : 64-col slab
  const int bm   = blockIdx.y * 128;
  const int bn   = blockIdx.x * 128;

  const int l16  = lane & 15;
  const int half = lane >> 4;
  const int kh   = half * 8;

  v8f zero = {};
  v8f acc[2][4];
#pragma unroll
  for (int i = 0; i < 2; ++i)
#pragma unroll
    for (int j = 0; j < 4; ++j) acc[i][j] = zero;

  const int ldRow = tid >> 1;            // 0..127
  const int ldCol = (tid & 1) * 16;      // 0 or 16

  // Stage one 128x32 bf16 tile of A and Bt into LDS buffer p.
  // Each thread moves 32B per array (two b128); INST_OFFSET applies to both
  // the global and LDS address, so the second chunk is offset:16.
  auto stage = [&](int p, int k0) {
    const unsigned short* ga = &A[(size_t)(bm + ldRow) * K + k0 + ldCol];
    const unsigned short* gb = &Bt[(size_t)(bn + ldRow) * K + k0 + ldCol];
    unsigned short* la = &sA[p][ldRow * 32 + ldCol];
    unsigned short* lb = &sB[p][ldRow * 32 + ldCol];
#if HAVE_ASYNC_LDS
    __builtin_amdgcn_global_load_async_to_lds_b128(as_glb_v4(ga), as_lds_v4(la), 0, 0);
    __builtin_amdgcn_global_load_async_to_lds_b128(as_glb_v4(ga), as_lds_v4(la), 16, 0);
    __builtin_amdgcn_global_load_async_to_lds_b128(as_glb_v4(gb), as_lds_v4(lb), 0, 0);
    __builtin_amdgcn_global_load_async_to_lds_b128(as_glb_v4(gb), as_lds_v4(lb), 16, 0);
#else
    *(u32x4*)la       = *(const u32x4*)ga;
    *(u32x4*)(la + 8) = *(const u32x4*)(ga + 8);
    *(u32x4*)lb       = *(const u32x4*)gb;
    *(u32x4*)(lb + 8) = *(const u32x4*)(gb + 8);
#endif
  };
  auto wait_async = [&]() {
#if HAVE_ASYNC_LDS
    __builtin_amdgcn_s_wait_asynccnt(0);
#endif
  };

  // Prologue: stage tile 0, make it visible.
  stage(0, 0);
  wait_async();
  __syncthreads();

  int p = 0;
  for (int k0 = 0; k0 < K; k0 += 32) {
    // Kick off next tile into the other buffer while we compute this one.
    if (k0 + 32 < K) stage(p ^ 1, k0 + 32);

    v16bf afrag[2], bfrag[4];
#pragma unroll
    for (int i = 0; i < 2; ++i) {
      const unsigned short* r = &sA[p][(wm * 32 + i * 16 + l16) * 32];
      afrag[i] = load_frag16(r + kh, r + 16 + kh);
    }
#pragma unroll
    for (int j = 0; j < 4; ++j) {
      const unsigned short* r = &sB[p][(wn * 64 + j * 16 + l16) * 32];
      bfrag[j] = load_frag16(r + kh, r + 16 + kh);
    }
#pragma unroll
    for (int i = 0; i < 2; ++i)
#pragma unroll
      for (int j = 0; j < 4; ++j)
        acc[i][j] = __builtin_amdgcn_wmma_f32_16x16x32_bf16(
            false, afrag[i], false, bfrag[j], (short)0, acc[i][j], false, false);

    // Next tile must have landed, and all waves be done with buffer p.
    wait_async();
    __syncthreads();
    p ^= 1;
  }

  // C layout: lane l holds n = l&15 ; VGPR r holds m = r + 8*(l>>4)
#pragma unroll
  for (int i = 0; i < 2; ++i)
#pragma unroll
    for (int j = 0; j < 4; ++j)
#pragma unroll
      for (int r = 0; r < 8; ++r) {
        int gm = bm + wm * 32 + i * 16 + r + 8 * half;
        int gn = bn + wn * 64 + j * 16 + l16;
        if (NT_STORE)
          __builtin_nontemporal_store(acc[i][j][r], &C[(size_t)gm * N + gn]);
        else
          C[(size_t)gm * N + gn] = acc[i][j][r];
      }
}

// ---------------------------------------------------------------------------
// RoPE + repack: in fp32 (B,S,nh*HD) -> out bf16 (B,nh,S,HD), folding an
// optional scale (1/sqrt(HD) for Q) into the values.
// ---------------------------------------------------------------------------
__global__ void rope_pack_kernel(const float* __restrict__ qk,
                                 const float* __restrict__ cosT,
                                 const float* __restrict__ sinT,
                                 unsigned short* __restrict__ out,
                                 int nh, float scale) {
  size_t idx = (size_t)blockIdx.x * blockDim.x + threadIdx.x;
  size_t total = (size_t)B_ * nh * S_ * HD_;
  if (idx >= total) return;
  int hd = (int)(idx & (HD_ - 1));
  size_t t = idx >> 6;               // HD_ == 64
  int s = (int)(t % S_); t /= S_;
  int h = (int)(t % nh);
  int b = (int)(t / nh);
  const float* row = qk + ((size_t)b * S_ + s) * ((size_t)nh * HD_) + (size_t)h * HD_;
  float x  = row[hd];
  float c  = cosT[(size_t)s * HD_ + hd];
  float sn = sinT[(size_t)s * HD_ + hd];
  float rot = (hd < HD_ / 2) ? -row[hd + HD_ / 2] : row[hd - HD_ / 2];
  out[idx] = f2bf((x * c + rot * sn) * scale);
}

// ---------------------------------------------------------------------------
// V repack/transpose: in fp32 (B,S,G*HD) -> out bf16 (B,G,HD,S) so the
// P*V WMMA B-operand (columns of V) is contiguous.
// ---------------------------------------------------------------------------
__global__ void vpack_kernel(const float* __restrict__ v,
                             unsigned short* __restrict__ vt) {
  size_t idx = (size_t)blockIdx.x * blockDim.x + threadIdx.x;
  size_t total = (size_t)B_ * G_ * HD_ * S_;
  if (idx >= total) return;
  int s = (int)(idx % S_); size_t t = idx / S_;
  int hd = (int)(t % HD_); t /= HD_;
  int g = (int)(t % G_);
  int b = (int)(t / G_);
  vt[idx] = f2bf(v[((size_t)b * S_ + s) * (G_ * HD_) + (size_t)g * HD_ + hd]);
}

// ---------------------------------------------------------------------------
// Flash attention (causal, GQA). 256 threads = 8 waves; each wave owns a
// 16-row Q tile (block covers 128 rows). KV tiles of 32. Scores and P*V go
// through v_wmma_f32_16x16x32_bf16; online softmax in registers with
// shfl_xor butterflies confined to the 16-lane halves, which keeps the
// per-row stats aligned with the C-layout row mapping (m = r + 8*half).
// ---------------------------------------------------------------------------
__global__ __launch_bounds__(256) void attn_kernel(
    const unsigned short* __restrict__ qbf,   // (B,H,S,HD)
    const unsigned short* __restrict__ kbf,   // (B,G,S,HD)
    const unsigned short* __restrict__ vtbf,  // (B,G,HD,S)
    unsigned short* __restrict__ ctx) {       // (B,S,H*HD) bf16
  __shared__ unsigned short pbuf[8][16 * 32]; // per-wave P tile (16q x 32kv)

  const int tid  = threadIdx.x;
  const int lane = tid & 31;
  const int wid  = tid >> 5;
  const int l16  = lane & 15;
  const int half = lane >> 4;
  const int kh   = half * 8;

  const int b  = blockIdx.z;
  const int h  = blockIdx.y;
  const int g  = h / (H_ / G_);
  const int q0 = blockIdx.x * 128 + wid * 16;

  const unsigned short* qbase = qbf  + (((size_t)b * H_ + h) * S_) * HD_;
  const unsigned short* kbase = kbf  + (((size_t)b * G_ + g) * S_) * HD_;
  const unsigned short* vbase = vtbf + (((size_t)b * G_ + g) * HD_) * S_;

  // Q fragments: row q0 + l16, HD split into two K=32 steps. Held all kernel.
  v16bf qfrag[2];
  {
    const unsigned short* r = qbase + (size_t)(q0 + l16) * HD_;
    qfrag[0] = load_frag16(r + kh,      r + 16 + kh);
    qfrag[1] = load_frag16(r + 32 + kh, r + 48 + kh);
  }

  v8f zero = {};
  v8f cacc[4];                       // ctx accum: 4 HD-subtiles of 16
#pragma unroll
  for (int j = 0; j < 4; ++j) cacc[j] = zero;
  float mrun[8], lrun[8];
#pragma unroll
  for (int r = 0; r < 8; ++r) { mrun[r] = -3.0e38f; lrun[r] = 0.0f; }

  const int qmax = q0 + 15;
  for (int kv0 = 0; kv0 <= qmax; kv0 += 32) {
    if (kv0 + 32 <= qmax) {
      // hint next K tile toward the caches (global_prefetch_b8)
      __builtin_prefetch(kbase + (size_t)(kv0 + 32 + l16) * HD_, 0, 0);
    }

    // ---- scores: S = Q * K^T (K rows are the B^T fragments) ----
    v8f sacc[2] = {zero, zero};
#pragma unroll
    for (int sub = 0; sub < 2; ++sub) {
      const unsigned short* kr = kbase + (size_t)(kv0 + sub * 16 + l16) * HD_;
      v16bf kf0 = load_frag16(kr + kh,      kr + 16 + kh);
      v16bf kf1 = load_frag16(kr + 32 + kh, kr + 48 + kh);
      sacc[sub] = __builtin_amdgcn_wmma_f32_16x16x32_bf16(
          false, qfrag[0], false, kf0, (short)0, sacc[sub], false, false);
      sacc[sub] = __builtin_amdgcn_wmma_f32_16x16x32_bf16(
          false, qfrag[1], false, kf1, (short)0, sacc[sub], false, false);
    }

    // ---- causal mask + online softmax; write P tile to LDS ----
#pragma unroll
    for (int r = 0; r < 8; ++r) {
      const int qg = q0 + r + 8 * half;          // global q row of this slot
      float s0 = sacc[0][r];
      float s1 = sacc[1][r];
      if (kv0 + l16      > qg) s0 = -3.0e38f;
      if (kv0 + 16 + l16 > qg) s1 = -3.0e38f;

      float mt = fmaxf(s0, s1);
#pragma unroll
      for (int msk = 1; msk <= 8; msk <<= 1)
        mt = fmaxf(mt, __shfl_xor(mt, msk, 32));
      float mnew = fmaxf(mrun[r], mt);

      float p0 = __expf(s0 - mnew);
      float p1 = __expf(s1 - mnew);
      float rs = p0 + p1;
#pragma unroll
      for (int msk = 1; msk <= 8; msk <<= 1)
        rs += __shfl_xor(rs, msk, 32);

      float sc = __expf(mrun[r] - mnew);
      mrun[r] = mnew;
      lrun[r] = lrun[r] * sc + rs;
#pragma unroll
      for (int j = 0; j < 4; ++j) cacc[j][r] *= sc;

      const int qloc = r + 8 * half;
      pbuf[wid][qloc * 32 + l16]      = f2bf(p0);
      pbuf[wid][qloc * 32 + 16 + l16] = f2bf(p1);
    }

    // LDS ops from one wave complete in order (DScnt); stop the compiler
    // from reordering across the C-layout -> A-layout handoff.
    __builtin_amdgcn_wave_barrier();
    asm volatile("" ::: "memory");

    // ---- ctx += P * V ----
    v16bf pfrag;
    {
      const unsigned short* pr = &pbuf[wid][l16 * 32];
      pfrag = load_frag16(pr + kh, pr + 16 + kh);
    }
#pragma unroll
    for (int j = 0; j < 4; ++j) {
      const unsigned short* vr = vbase + (size_t)(j * 16 + l16) * S_ + kv0;
      v16bf vf = load_frag16(vr + kh, vr + 16 + kh);
      cacc[j] = __builtin_amdgcn_wmma_f32_16x16x32_bf16(
          false, pfrag, false, vf, (short)0, cacc[j], false, false);
    }
    __builtin_amdgcn_wave_barrier();
    asm volatile("" ::: "memory");
  }

  // ---- normalize and write ctx (bf16, (B,S,H*HD)) ----
#pragma unroll
  for (int r = 0; r < 8; ++r) {
    const float inv = 1.0f / lrun[r];
    const int qg = q0 + r + 8 * half;
    unsigned short* orow =
        ctx + ((size_t)b * S_ + qg) * ((size_t)H_ * HD_) + (size_t)h * HD_;
#pragma unroll
    for (int j = 0; j < 4; ++j)
      orow[j * 16 + l16] = f2bf(cacc[j][r] * inv);
  }
}

// ---------------------------------------------------------------------------
// Host-side orchestration (graph-capture safe: only kernel launches).
// Inputs: 0:x 1:attn_mask(unused) 2:cos 3:sin 4:Wq 5:Wk 6:Wv 7:Wo
// ---------------------------------------------------------------------------
extern "C" void kernel_launch(void* const* d_in, const int* in_sizes, int n_in,
                              void* d_out, int out_size, void* d_ws, size_t ws_size,
                              hipStream_t stream) {
  (void)in_sizes; (void)n_in; (void)out_size; (void)ws_size;
  const float* x    = (const float*)d_in[0];
  const float* cosT = (const float*)d_in[2];
  const float* sinT = (const float*)d_in[3];
  const float* Wq   = (const float*)d_in[4];
  const float* Wk   = (const float*)d_in[5];
  const float* Wv   = (const float*)d_in[6];
  const float* Wo   = (const float*)d_in[7];
  float* out = (float*)d_out;

  char* ws = (char*)d_ws;
  size_t off = 0;
  auto alloc = [&](size_t bytes) -> char* {
    char* p = ws + off;
    off = (off + bytes + 255) & ~(size_t)255;
    return p;
  };

  const size_t MS = (size_t)B_ * S_;        // 4096 rows
  unsigned short* xbf = (unsigned short*)alloc(MS * D_ * 2);
  unsigned short* wqT = (unsigned short*)alloc((size_t)(H_ * HD_) * D_ * 2);
  unsigned short* wkT = (unsigned short*)alloc((size_t)(G_ * HD_) * D_ * 2);
  unsigned short* wvT = (unsigned short*)alloc((size_t)(G_ * HD_) * D_ * 2);
  unsigned short* woT = (unsigned short*)alloc((size_t)D_ * (H_ * HD_) * 2);
  float* qf = (float*)alloc(MS * (H_ * HD_) * 4);
  float* kf = (float*)alloc(MS * (G_ * HD_) * 4);
  float* vf = (float*)alloc(MS * (G_ * HD_) * 4);
  unsigned short* qpk = (unsigned short*)alloc((size_t)B_ * H_ * S_ * HD_ * 2);
  unsigned short* kpk = (unsigned short*)alloc((size_t)B_ * G_ * S_ * HD_ * 2);
  unsigned short* vtp = (unsigned short*)alloc((size_t)B_ * G_ * HD_ * S_ * 2);
  unsigned short* ctx = (unsigned short*)qf;   // reuse qf (consumed before attn)

  // 1) cast x -> bf16
  {
    size_t n = MS * D_;
    cast_f32_bf16_kernel<<<dim3((unsigned)((n + 255) / 256)), 256, 0, stream>>>(x, xbf, n);
  }

  // 2) transpose-cast weights to B^T (N x K) bf16
  transpose_cast_kernel<<<dim3((H_ * HD_) / 32, D_ / 32), dim3(32, 8), 0, stream>>>(Wq, wqT, D_, H_ * HD_);
  transpose_cast_kernel<<<dim3((G_ * HD_) / 32, D_ / 32), dim3(32, 8), 0, stream>>>(Wk, wkT, D_, G_ * HD_);
  transpose_cast_kernel<<<dim3((G_ * HD_) / 32, D_ / 32), dim3(32, 8), 0, stream>>>(Wv, wvT, D_, G_ * HD_);
  transpose_cast_kernel<<<dim3(D_ / 32, (H_ * HD_) / 32), dim3(32, 8), 0, stream>>>(Wo, woT, H_ * HD_, D_);

  // 3) QKV projections (WMMA GEMM, fp32 accum out; results are re-read -> RT stores)
  gemm_bf16_kernel<false><<<dim3((H_ * HD_) / 128, (unsigned)(MS / 128)), 256, 0, stream>>>(xbf, wqT, qf, (int)MS, H_ * HD_, D_);
  gemm_bf16_kernel<false><<<dim3((G_ * HD_) / 128, (unsigned)(MS / 128)), 256, 0, stream>>>(xbf, wkT, kf, (int)MS, G_ * HD_, D_);
  gemm_bf16_kernel<false><<<dim3((G_ * HD_) / 128, (unsigned)(MS / 128)), 256, 0, stream>>>(xbf, wvT, vf, (int)MS, G_ * HD_, D_);

  // 4) RoPE + repack (fold 1/sqrt(HD) into Q); V transpose-pack
  {
    size_t nq = (size_t)B_ * H_ * S_ * HD_;
    rope_pack_kernel<<<dim3((unsigned)((nq + 255) / 256)), 256, 0, stream>>>(qf, cosT, sinT, qpk, H_, 0.125f);
    size_t nk = (size_t)B_ * G_ * S_ * HD_;
    rope_pack_kernel<<<dim3((unsigned)((nk + 255) / 256)), 256, 0, stream>>>(kf, cosT, sinT, kpk, G_, 1.0f);
    vpack_kernel<<<dim3((unsigned)((nk + 255) / 256)), 256, 0, stream>>>(vf, vtp);
  }

  // 5) causal flash attention -> ctx bf16 (B,S,H*HD)
  attn_kernel<<<dim3(S_ / 128, H_, B_), 256, 0, stream>>>(qpk, kpk, vtp, ctx);

  // 6) output projection -> d_out fp32 (write-once -> non-temporal stores)
  gemm_bf16_kernel<true><<<dim3(D_ / 128, (unsigned)(MS / 128)), 256, 0, stream>>>(ctx, woT, out, (int)MS, D_, H_ * HD_);
}
